// GIBGCN_4071628996670
// MI455X (gfx1250) — matrix operand order repaired and checked by reading, not compiled, not run on hardware
//
#include <hip/hip_runtime.h>
#include <hip/hip_bf16.h>
#include <math.h>

// ---------------------------------------------------------------------------
// GIB-GCN on MI455X (gfx1250, wave32).
//   N=50000, F=512, LATENT=64, CLASSES=32, E=800000, S=1
//
// Roofline: GEMM work (~7 GFLOP) is negligible for a WMMA-class machine; the
// bound is ~1.3GB of gather/scatter traffic + 109M f32 atomics, all resident
// in the 192MB L2 (proj1+agg1 = 51MB). Strategy:
//   * GEMMs: one wave32 per 16-row strip, v_wmma_f32_16x16x32_f16, A loaded
//     once per K-step and reused across all N-tiles; W pre-converted to a
//     transposed f16 layout so each B fragment is ONE contiguous 32B load.
//   * Aggregation: edge-per-wave coalesced gather + atomicAdd scatter (L2).
//   * Finalize: wave-per-node, lane-parallel features, __shfl_xor reduction.
// ---------------------------------------------------------------------------

typedef __attribute__((ext_vector_type(16))) _Float16 v16h;
typedef __attribute__((ext_vector_type(8)))  float    v8f;

// ---------------- degree / normalization ----------------------------------

__global__ void k_init_deg(float* __restrict__ deg, int n) {
    int i = blockIdx.x * blockDim.x + threadIdx.x;
    if (i < n) deg[i] = 1.0f;   // self-loop contributes 1 to every node
}

__global__ void k_accum_deg(const int* __restrict__ col,
                            const float* __restrict__ ew,
                            float* __restrict__ deg, int e) {
    int i = blockIdx.x * blockDim.x + threadIdx.x;
    if (i < e) atomicAdd(&deg[col[i]], ew[i]);
}

__global__ void k_dinv(const float* __restrict__ deg,
                       float* __restrict__ dinv, int n) {
    int i = blockIdx.x * blockDim.x + threadIdx.x;
    if (i < n) {
        float d = deg[i];
        dinv[i] = (d > 0.0f) ? rsqrtf(fmaxf(d, 1e-30f)) : 0.0f;
    }
}

// Wt[c][k] = (f16) W[k][c]  -- transposed f16 weights so B fragments are
// contiguous 32B per lane in the WMMA B layout.
__global__ void k_convW(const float* __restrict__ W, _Float16* __restrict__ Wt,
                        int K, int Cn) {
    int t = blockIdx.x * blockDim.x + threadIdx.x;
    if (t >= K * Cn) return;
    int c = t / K;
    int k = t - c * K;
    Wt[t] = (_Float16)W[k * Cn + c];
}

// ---------------- WMMA GEMM: P[M x Cn] = X[M x K] @ W[K x Cn] --------------
// One wave32 computes a full 16 x Cn strip (Cn/16 accumulators), stepping K
// by 32 with v_wmma_f32_16x16x32_f16. Fragment packing used here
// (contiguous k order, same in A and B; not AMD's table order):
//   A (16x32 f16): lane row = lane&15; K offsets h + 16*(lane>=16).
//   B (32x16 f16): lanes 0-15 K=0..15, lanes 16-31 K=16..31; N=lane&15.
//     With Wt[Cn][K] f16 row-major, lane's 16 halves are contiguous -> b128x2.
//   C/D (16x16 f32): VGPR r -> M = r + 8*(lane>=16), N = lane&15.
// Requires M%16==0, Cn%16==0, K%32==0 (true: 50000, 128/64, 512/64).
template <int Cn, int K>
__global__ void k_gemm_wmma(const float* __restrict__ X,
                            const _Float16* __restrict__ Wt,
                            float* __restrict__ P, int M) {
    const int wave = threadIdx.x >> 5;
    const int lane = threadIdx.x & 31;
    const int strip = blockIdx.x * (blockDim.x >> 5) + wave;
    if (strip >= M / 16) return;      // wave-uniform: EXEC stays all-1s

    constexpr int TN = Cn / 16;
    const int m0 = strip * 16;
    const int hi = lane >> 4;         // 0: lanes 0-15, 1: lanes 16-31
    const int l15 = lane & 15;

    v8f acc[TN] = {};
    const float* xrow = X + (long long)(m0 + l15) * K + hi * 16;

    for (int k0 = 0; k0 < K; k0 += 32) {
        // A fragment: loaded once, reused for all TN column tiles.
        v16h a;
#pragma unroll
        for (int h = 0; h < 16; ++h) {
            int kk = h;
            a[h] = (_Float16)xrow[k0 + kk];
        }
        if (k0 + 32 < K)
            __builtin_prefetch(xrow + k0 + 32, 0, 3);  // global_prefetch_b8

#pragma unroll
        for (int tn = 0; tn < TN; ++tn) {
            // One contiguous 32B f16 load per lane (32B-aligned by layout).
            const v16h b = *(const v16h*)(Wt + (long long)(tn * 16 + l15) * K
                                             + k0 + hi * 16);
            acc[tn] = __builtin_amdgcn_wmma_f32_16x16x32_f16(
                /*neg_a=*/false, a, /*neg_b=*/false, b,
                /*c_mod=*/(short)0, acc[tn], /*reuse_a=*/false, /*reuse_b=*/false);
        }
    }

#pragma unroll
    for (int tn = 0; tn < TN; ++tn) {
#pragma unroll
        for (int r = 0; r < 8; ++r) {
            P[(long long)(m0 + r + hi * 8) * Cn + tn * 16 + l15] = acc[tn][r];
        }
    }
}

// ---------------- aggregation --------------------------------------------

// agg[i][j] = b[j]  (bias pre-seed, so the scatter just accumulates)
__global__ void k_init_agg(float* __restrict__ agg, const float* __restrict__ b,
                           int n, int log2C2) {
    long long t = (long long)blockIdx.x * blockDim.x + threadIdx.x;
    long long total = (long long)n << log2C2;
    if (t >= total) return;
    int j = (int)(t & ((1 << log2C2) - 1));
    agg[t] = b[j];
}

// One thread per (edge, feature). Threads within a wave share an edge and
// cover consecutive features -> coalesced gather reads and coalesced atomic
// adds resolved at the L2 atomic units (proj/agg are L2-resident).
// Edges [0,E) are real edges; [E, E+N) are self-loops (ew=1).
__global__ void k_scatter(const int* __restrict__ row,
                          const int* __restrict__ col,
                          const float* __restrict__ ew,
                          const float* __restrict__ dinv,
                          const float* __restrict__ proj,
                          float* __restrict__ agg,
                          int E, int n, int log2C) {
    long long t = (long long)blockIdx.x * blockDim.x + threadIdx.x;
    long long total = (long long)(E + n) << log2C;
    if (t >= total) return;
    int e = (int)(t >> log2C);
    int j = (int)(t & ((1 << log2C) - 1));
    int r, cl;
    float nrm;
    if (e < E) {
        r = row[e];
        cl = col[e];
        nrm = dinv[r] * ew[e] * dinv[cl];
    } else {
        r = cl = e - E;
        float d = dinv[r];
        nrm = d * d;
    }
    float v = proj[((long long)r << log2C) + j] * nrm;
    atomicAdd(&agg[((long long)cl << log2C) + j], v);
}

// ---------------- softplus / reparameterize / ixz -------------------------
// One wave32 per node: lanes cover the feature dim (coalesced), ixz reduced
// with a wave32 __shfl_xor butterfly.
template <int C>
__global__ void k_finalize(const float* __restrict__ agg,   // [n, 2C]
                           const float* __restrict__ eps,   // [1, n, C]
                           float* __restrict__ Z,           // [n, C]
                           float* __restrict__ ixz,         // [n]
                           float* __restrict__ skl,         // nullptr or [1]
                           int n) {
    const int wave = threadIdx.x >> 5;
    const int lane = threadIdx.x & 31;
    const int i = blockIdx.x * (blockDim.x >> 5) + wave;
    if (i >= n) return;

    constexpr int PER = C / 32;    // 2 for LATENT=64, 1 for CLASSES=32
    const float* arow = agg + (long long)i * (2 * C);
    const float* erow = eps + (long long)i * C;
    float* zrow = Z + (long long)i * C;

    float s = 0.0f;
#pragma unroll
    for (int p = 0; p < PER; ++p) {
        int j = p * 32 + lane;
        float mean = arow[j];
        float x = arow[C + j];
        float sp = (x > 20.0f) ? x : log1pf(__expf(x));   // softplus
        float sd = sp + 1e-10f;
        zrow[j] = mean + sd * erow[j];
        s += -__logf(sd) + 0.5f * (sd * sd + mean * mean) - 0.5f;
    }
#pragma unroll
    for (int off = 16; off > 0; off >>= 1)
        s += __shfl_xor(s, off, 32);
    if (lane == 0) ixz[i] = s;
    if (skl != nullptr && i == 0 && lane == 0) *skl = 0.0f;  // structure_kl
}

// ---------------------------------------------------------------------------

extern "C" void kernel_launch(void* const* d_in, const int* in_sizes, int n_in,
                              void* d_out, int out_size, void* d_ws, size_t ws_size,
                              hipStream_t stream) {
    const float* x    = (const float*)d_in[0];   // [N, 512]
    const int*   ei   = (const int*)  d_in[1];   // [2, E]
    const float* ew   = (const float*)d_in[2];   // [E]
    const float* W1   = (const float*)d_in[3];   // [512, 128]
    const float* b1   = (const float*)d_in[4];   // [128]
    const float* W2   = (const float*)d_in[5];   // [64, 64]
    const float* b2   = (const float*)d_in[6];   // [64]
    const float* eps1 = (const float*)d_in[7];   // [1, N, 64]
    const float* eps2 = (const float*)d_in[8];   // [1, N, 32]

    const int F  = 512;
    const int CL = 32;                 // CLASSES
    const int N  = in_sizes[0] / F;    // 50000 (multiple of 16)
    const int E  = in_sizes[2];        // 800000

    const int* rowIdx = ei;            // edge_index[0]
    const int* colIdx = ei + E;        // edge_index[1]

    // Workspace layout (floats; all region offsets are multiples of N floats,
    // N%16==0 so every region is 32B-aligned). proj/agg reused by layer 2.
    float* ws    = (float*)d_ws;
    float* deg   = ws;                               // N
    float* dinv  = ws + (size_t)N;                   // N
    float* proj  = ws + (size_t)2 * N;               // N*128 (L1) / N*64 (L2)
    float* agg   = proj + (size_t)N * 128;           // N*128 (L1) / N*64 (L2)
    float* h1    = agg + (size_t)N * 128;            // N*64
    _Float16* Wt1 = (_Float16*)(h1 + (size_t)N * 64);        // 512*128 halves
    _Float16* Wt2 = Wt1 + (size_t)512 * 128;                 // 64*64 halves

    // Output layout: h2 [N*32] | ixz1 [N] | ixz2 [N] | structure_kl [1]
    float* out  = (float*)d_out;
    float* h2   = out;
    float* ixz1 = out + (size_t)N * CL;
    float* ixz2 = ixz1 + N;
    float* skl  = ixz2 + N;

    const int B = 256;
    auto blocks = [](long long total, int b) { return (unsigned)((total + b - 1) / b); };

    // --- GCN normalization + weight conversion ---
    k_init_deg<<<blocks(N, B), B, 0, stream>>>(deg, N);
    k_accum_deg<<<blocks(E, B), B, 0, stream>>>(colIdx, ew, deg, E);
    k_dinv<<<blocks(N, B), B, 0, stream>>>(deg, dinv, N);
    k_convW<<<blocks(512 * 128, B), B, 0, stream>>>(W1, Wt1, 512, 128);
    k_convW<<<blocks(64 * 64, B), B, 0, stream>>>(W2, Wt2, 64, 64);

    // --- Layer 1: proj1 = x @ W1 (50000x512 @ 512x128) ---
    k_gemm_wmma<128, 512><<<blocks(N / 16, 8), B, 0, stream>>>(x, Wt1, proj, N);
    k_init_agg<<<blocks((long long)N * 128, B), B, 0, stream>>>(agg, b1, N, 7);
    k_scatter<<<blocks(((long long)E + N) * 128, B), B, 0, stream>>>(
        rowIdx, colIdx, ew, dinv, proj, agg, E, N, 7);
    k_finalize<64><<<blocks((long long)N * 32, B), B, 0, stream>>>(
        agg, eps1, h1, ixz1, nullptr, N);

    // --- Layer 2: proj2 = h1 @ W2 (50000x64 @ 64x64) ---
    k_gemm_wmma<64, 64><<<blocks(N / 16, 8), B, 0, stream>>>(h1, Wt2, proj, N);
    k_init_agg<<<blocks((long long)N * 64, B), B, 0, stream>>>(agg, b2, N, 6);
    k_scatter<<<blocks(((long long)E + N) * 64, B), B, 0, stream>>>(
        rowIdx, colIdx, ew, dinv, proj, agg, E, N, 6);
    k_finalize<32><<<blocks((long long)N * 32, B), B, 0, stream>>>(
        agg, eps2, h2, ixz2, skl, N);
}
